// GRUExponentialDecay_50517405335820
// MI455X (gfx1250) — compile-verified, matrix operand order checked by reading, not run on hardware
//
#include <hip/hip_runtime.h>

// Problem sizes (fixed by the reference)
#define B_  128
#define T_  256
#define I_  512
#define H_  1024
#define KC_ 128                  // K-chunk staged in LDS per double-buffer slot
#define NCH1_ (I_ / KC_)         // 4 chunks of the input projection
#define NCH2_ (H_ / KC_)         // 8 chunks of the hidden projection

typedef __attribute__((ext_vector_type(16))) __bf16        v16bf;
typedef __attribute__((ext_vector_type(8)))  float          v8f;
typedef __attribute__((ext_vector_type(8)))  unsigned short u16x8;

union FragAB {
    u16x8 h[2];   // two 16-byte chunks per lane
    v16bf v;      // WMMA operand view
};

__device__ __forceinline__ unsigned short f32_to_bf16_rne(float f) {
    union { float f; unsigned u; } v; v.f = f;
    unsigned r = v.u + 0x7FFFu + ((v.u >> 16) & 1u);  // round-to-nearest-even
    return (unsigned short)(r >> 16);
}

__device__ __forceinline__ float sigm(float x) {
    return 1.0f / (1.0f + __expf(-x));
}
__device__ __forceinline__ float tanh_fast(float x) {
    float e = __expf(2.0f * x);
    return (e - 1.0f) / (e + 1.0f);
}

// CDNA5 async copy: 16B global -> LDS per lane, tracked by ASYNCcnt.
__device__ __forceinline__ void async_ld_b128(unsigned lds_off, const void* gsrc) {
    asm volatile("global_load_async_to_lds_b128 %0, %1, off"
                 :: "v"(lds_off), "v"(gsrc) : "memory");
}
__device__ __forceinline__ void wait_async0() {
    asm volatile("s_wait_asynccnt 0x0" ::: "memory");
}

// ---------------------------------------------------------------------------
// Prep: f32 -> bf16 conversion (grid-stride)
// ---------------------------------------------------------------------------
__global__ void cvt_f32_bf16(const float* __restrict__ src,
                             unsigned short* __restrict__ dst, int n) {
    int i = blockIdx.x * blockDim.x + threadIdx.x;
    int stride = gridDim.x * blockDim.x;
    for (; i < n; i += stride) dst[i] = f32_to_bf16_rne(src[i]);
}

// Prep: zero initial hidden state (f32 + bf16 copies)
__global__ void init_h(float* __restrict__ hf,
                       unsigned short* __restrict__ hbf, int n) {
    int i = blockIdx.x * blockDim.x + threadIdx.x;
    int stride = gridDim.x * blockDim.x;
    for (; i < n; i += stride) { hf[i] = 0.0f; hbf[i] = 0; }
}

// ---------------------------------------------------------------------------
// Stage one weight chunk (48 rows = 3 gates x 16 cols, KC_ K-values) into LDS
// with async global->LDS b128 copies. 768 16B transfers / 256 threads = 3 each.
// LDS layout: [48][KC_] bf16, row = gate*16 + col-within-tile.
// ---------------------------------------------------------------------------
__device__ __forceinline__ void stage48(const unsigned short* __restrict__ W,
                                        int ldw, int colbase, int kbase,
                                        unsigned short* dst_lds, int tid) {
    const unsigned base = (unsigned)(size_t)dst_lds;   // LDS byte offset
    #pragma unroll
    for (int i = 0; i < 3; ++i) {
        const int c   = tid + i * 256;   // [0, 768)
        const int r48 = c >> 4;          // staged row in [0, 48)
        const int kp  = (c & 15) << 3;   // k offset within chunk (halves)
        const int g   = r48 >> 4;        // gate
        const int cc  = r48 & 15;        // column within tile
        const unsigned short* src =
            W + (size_t)(g * H_ + colbase + cc) * ldw + (kbase + kp);
        async_ld_b128(base + (unsigned)((r48 * KC_ + kp) * 2), src);
    }
}

// ---------------------------------------------------------------------------
// One GRU time step. Block = 256 threads = 8 waves; block owns one 16-column
// tile (all 3 gates) and ALL 8 row tiles -> weights staged once per block in
// LDS (double-buffered, async) and consumed by every wave via ds_load_b128.
//   acc_r  = x@Wr^T + h@Ur^T   (fused)     acc_z = x@Wz^T + h@Uz^T (fused)
//   acc_ni = x@Wn^T            acc_nh = h@Un^T   (r applies to acc_nh only)
// ---------------------------------------------------------------------------
__global__ __launch_bounds__(256, 2)
void gru_step(const unsigned short* __restrict__ xbf,   // [B,T,I] bf16
              const unsigned short* __restrict__ wih,   // [3H,I]  bf16
              const unsigned short* __restrict__ whh,   // [3H,H]  bf16
              const float* __restrict__ b_ih,           // [3H]
              const float* __restrict__ b_hh,           // [3H]
              const float* __restrict__ times,          // [B,T]
              const float* __restrict__ decays,         // scalar
              const float* __restrict__ h_prev_f,       // [B,H] f32
              const unsigned short* __restrict__ h_prev_bf, // [B,H] bf16
              float* __restrict__ h_next_f,             // [B,H] f32
              unsigned short* __restrict__ h_next_bf,   // [B,H] bf16
              float* __restrict__ out,                  // [B,T,H]
              int t) {
    __shared__ unsigned short lbuf[2][48 * KC_];        // 2 x 12 KB

    const int tid  = threadIdx.x;
    const int lane = tid & 31;
    const int wv   = tid >> 5;          // wave id = row tile (B = 8 x 16)
    const int rowbase = wv * 16;
    const int colbase = blockIdx.x * 16;  // 64 blocks cover H = 1024
    const int half = lane >> 4;         // K-half selector (ISA bf16 layout)
    const int r16  = lane & 15;         // row (A) / column (B) within tile
    const int j    = colbase + r16;     // output column owned by this lane

    v8f acc_r  = {};
    v8f acc_z  = {};
    v8f acc_ni = {};
    v8f acc_nh = {};

    const unsigned short* axrow = xbf + ((size_t)(rowbase + r16) * T_ + t) * I_;
    const unsigned short* ahrow = h_prev_bf + (size_t)(rowbase + r16) * H_;

    // Prologue: stage chunk 0 of W_ih.
    stage48(wih, I_, colbase, 0, lbuf[0], tid);
    wait_async0();
    __syncthreads();

    int buf = 0;
    for (int ch = 0; ch < NCH1_ + NCH2_; ++ch) {
        // Kick off async staging of the next chunk into the other buffer.
        const int nxt = ch + 1;
        if (nxt < NCH1_) {
            stage48(wih, I_, colbase, nxt * KC_, lbuf[buf ^ 1], tid);
        } else if (nxt < NCH1_ + NCH2_) {
            stage48(whh, H_, colbase, (nxt - NCH1_) * KC_, lbuf[buf ^ 1], tid);
        }

        // Compute on the current chunk: 4 k-steps x 3 WMMAs.
        const unsigned short* l0 = &lbuf[buf][( 0 + r16) * KC_];  // r gate row
        const unsigned short* l1 = &lbuf[buf][(16 + r16) * KC_];  // z gate row
        const unsigned short* l2 = &lbuf[buf][(32 + r16) * KC_];  // n gate row
        if (ch < NCH1_) {
            const unsigned short* arow = axrow + ch * KC_;
            #pragma unroll
            for (int kk = 0; kk < KC_; kk += 32) {
                FragAB a, fr, fz, fn;
                a.h[0]  = *(const u16x8*)(arow + kk + half * 8);
                a.h[1]  = *(const u16x8*)(arow + kk + 16 + half * 8);
                fr.h[0] = *(const u16x8*)(l0 + kk + half * 8);
                fr.h[1] = *(const u16x8*)(l0 + kk + 16 + half * 8);
                fz.h[0] = *(const u16x8*)(l1 + kk + half * 8);
                fz.h[1] = *(const u16x8*)(l1 + kk + 16 + half * 8);
                fn.h[0] = *(const u16x8*)(l2 + kk + half * 8);
                fn.h[1] = *(const u16x8*)(l2 + kk + 16 + half * 8);
                acc_r  = __builtin_amdgcn_wmma_f32_16x16x32_bf16(
                             false, a.v, false, fr.v, (short)0, acc_r, false, false);
                acc_z  = __builtin_amdgcn_wmma_f32_16x16x32_bf16(
                             false, a.v, false, fz.v, (short)0, acc_z, false, false);
                acc_ni = __builtin_amdgcn_wmma_f32_16x16x32_bf16(
                             false, a.v, false, fn.v, (short)0, acc_ni, false, false);
            }
        } else {
            const unsigned short* arow = ahrow + (ch - NCH1_) * KC_;
            #pragma unroll
            for (int kk = 0; kk < KC_; kk += 32) {
                FragAB a, fr, fz, fn;
                a.h[0]  = *(const u16x8*)(arow + kk + half * 8);
                a.h[1]  = *(const u16x8*)(arow + kk + 16 + half * 8);
                fr.h[0] = *(const u16x8*)(l0 + kk + half * 8);
                fr.h[1] = *(const u16x8*)(l0 + kk + 16 + half * 8);
                fz.h[0] = *(const u16x8*)(l1 + kk + half * 8);
                fz.h[1] = *(const u16x8*)(l1 + kk + 16 + half * 8);
                fn.h[0] = *(const u16x8*)(l2 + kk + half * 8);
                fn.h[1] = *(const u16x8*)(l2 + kk + 16 + half * 8);
                acc_r  = __builtin_amdgcn_wmma_f32_16x16x32_bf16(
                             false, a.v, false, fr.v, (short)0, acc_r, false, false);
                acc_z  = __builtin_amdgcn_wmma_f32_16x16x32_bf16(
                             false, a.v, false, fz.v, (short)0, acc_z, false, false);
                acc_nh = __builtin_amdgcn_wmma_f32_16x16x32_bf16(
                             false, a.v, false, fn.v, (short)0, acc_nh, false, false);
            }
        }

        // Staged chunk complete + all waves done with current buffer.
        wait_async0();
        __syncthreads();
        buf ^= 1;
    }

    // ---- Epilogue: gates + decay ----------------------------------------
    // C/D layout: element v of v8f is (M = v + 8*half, N = r16) of the tile.
    const float biasr = b_ih[j] + b_hh[j];
    const float biasz = b_ih[H_ + j] + b_hh[H_ + j];
    const float bin   = b_ih[2 * H_ + j];
    const float bhn   = b_hh[2 * H_ + j];
    const float dk    = decays[0];

    #pragma unroll
    for (int v = 0; v < 8; ++v) {
        const int m = v + half * 8;
        const int b = rowbase + m;
        const float rg = sigm(acc_r[v] + biasr);
        const float zg = sigm(acc_z[v] + biasz);
        const float ng = tanh_fast(acc_ni[v] + bin + rg * (acc_nh[v] + bhn));
        const float hp = h_prev_f[(size_t)b * H_ + j];
        const float hn = (1.0f - zg) * ng + zg * hp;
        out[((size_t)b * T_ + t) * H_ + j] = hn;           // output: pre-decay
        const float tt  = times[(size_t)b * T_ + t];
        const float dec = __expf(-fmaxf(tt * dk, 0.0f));
        const float hc  = hn * dec;                        // carried state
        h_next_f[(size_t)b * H_ + j]  = hc;
        h_next_bf[(size_t)b * H_ + j] = f32_to_bf16_rne(hc);
    }
}

// ---------------------------------------------------------------------------
// Host launcher
// ---------------------------------------------------------------------------
extern "C" void kernel_launch(void* const* d_in, const int* in_sizes, int n_in,
                              void* d_out, int out_size, void* d_ws, size_t ws_size,
                              hipStream_t stream) {
    const float* x      = (const float*)d_in[0];  // [B,T,I]
    const float* times  = (const float*)d_in[1];  // [B,T]
    const float* w_ih   = (const float*)d_in[2];  // [3H,I]
    const float* w_hh   = (const float*)d_in[3];  // [3H,H]
    const float* b_ih   = (const float*)d_in[4];  // [3H]
    const float* b_hh   = (const float*)d_in[5];  // [3H]
    const float* decays = (const float*)d_in[6];  // scalar
    float* out = (float*)d_out;

    // Workspace layout (~45 MB total)
    char* ws = (char*)d_ws;
    size_t off = 0;
    unsigned short* xbf   = (unsigned short*)(ws + off); off += (size_t)B_ * T_ * I_ * 2;
    unsigned short* wihbf = (unsigned short*)(ws + off); off += (size_t)3 * H_ * I_ * 2;
    unsigned short* whhbf = (unsigned short*)(ws + off); off += (size_t)3 * H_ * H_ * 2;
    float*          hf    = (float*)(ws + off);          off += (size_t)2 * B_ * H_ * 4;
    unsigned short* hbf   = (unsigned short*)(ws + off); off += (size_t)2 * B_ * H_ * 2;
    (void)ws_size; (void)in_sizes; (void)n_in; (void)out_size;

    // Prep passes (fully parallel, HBM-bound)
    cvt_f32_bf16<<<2048, 256, 0, stream>>>(x,    xbf,   B_ * T_ * I_);
    cvt_f32_bf16<<<1024, 256, 0, stream>>>(w_ih, wihbf, 3 * H_ * I_);
    cvt_f32_bf16<<<1024, 256, 0, stream>>>(w_hh, whhbf, 3 * H_ * H_);
    init_h<<<256, 256, 0, stream>>>(hf, hbf, B_ * H_);

    // Sequential recurrence: 256 dependent step kernels.
    // 64 blocks x 8 waves; each block = one 16-col tile, weights via async LDS.
    for (int t = 0; t < T_; ++t) {
        const int pb = t & 1;
        const int nb = (t + 1) & 1;
        gru_step<<<64, 256, 0, stream>>>(
            xbf, wihbf, whhbf, b_ih, b_hh, times, decays,
            hf  + (size_t)pb * B_ * H_, hbf + (size_t)pb * B_ * H_,
            hf  + (size_t)nb * B_ * H_, hbf + (size_t)nb * B_ * H_,
            out, t);
    }
}
